// RGSACausalSelfAttention_68693706932453
// MI455X (gfx1250) — compile-verified, matrix operand order checked by reading, not run on hardware
//
#include <hip/hip_runtime.h>
#include <hip/hip_bf16.h>

typedef _Float16 v16h __attribute__((ext_vector_type(16)));
typedef _Float16 v8h  __attribute__((ext_vector_type(8)));
typedef float    v8f  __attribute__((ext_vector_type(8)));

#define T_SEQ   2048
#define C_DIM   768
#define NHEAD   12
#define HD      64
#define NCHUNK  32
#define B_SZ    2
#define M_ROWS  (B_SZ * T_SEQ)   // 4096
#define N3C     (3 * C_DIM)      // 2304
#define LWIN    256

// ---------------------------------------------------------------------------
// WMMA helpers (CDNA5 gfx1250, wave32)
// ---------------------------------------------------------------------------
__device__ __forceinline__ v8f wmma_f16(v16h a, v16h b, v8f c) {
  // D = A(16x32 f16) * B(32x16 f16) + C(16x16 f32)
  return __builtin_amdgcn_wmma_f32_16x16x32_f16(false, a, false, b, (short)0, c,
                                                false, false);
}

// Load a 16x32 WMMA A/B fragment from a matrix stored so element (row,k) is
// at base[row*ld + k].  Per the CDNA5 ISA 16-bit fragment layout:
//   lanes 0-15 : row = lane,    K = {0..7, 16..23}
//   lanes 16-31: row = lane-16, K = {8..15, 24..31}
// Both K groups are 8 contiguous halves -> two 16-byte loads per lane, which
// start 32 bytes apart, so they cannot merge into one 128-bit load.
__device__ __forceinline__ v16h load_frag(const _Float16* base, int ld,
                                          int row0, int k0, int lane) {
  const int r  = row0 + (lane & 15);
  const int kb = k0 + ((lane >> 4) << 3);
  const _Float16* p = base + (size_t)r * ld + kb;
  v8h lo = *(const v8h*)(p);        // K offsets kb .. kb+7
  v8h hi = *(const v8h*)(p + 16);   // K offsets kb+16 .. kb+23
  return __builtin_shufflevector(lo, hi, 0, 1, 2, 3, 4, 5, 6, 7,
                                 8, 9, 10, 11, 12, 13, 14, 15);
}

// ---------------------------------------------------------------------------
// Elementwise casts
// ---------------------------------------------------------------------------
__global__ void k_cast_f16(const float* __restrict__ in,
                           _Float16* __restrict__ out, int n) {
  int i = blockIdx.x * blockDim.x + threadIdx.x;
  if (i < n) out[i] = (_Float16)in[i];
}

// out[n*K + k] = (f16) in[k*N + n]   (cast + transpose so WMMA B-fragments
// read contiguously along K)
__global__ void k_cast_transpose(const float* __restrict__ in,
                                 _Float16* __restrict__ out, int K, int N) {
  int i = blockIdx.x * blockDim.x + threadIdx.x;
  if (i >= K * N) return;
  int n = i / K;
  int k = i - n * K;
  out[i] = (_Float16)in[(size_t)k * N + n];
}

// ---------------------------------------------------------------------------
// QKV projection: xh[4096,768] @ WaT'[2304,768] + b_attn
//   -> q,k : [B,H,T,64] f16 (row-major over hd)
//   -> v   : [B,H,64,T] f16 (TRANSPOSED: so attention's P*V B-fragments read
//            contiguously along the key dimension straight from global)
// 32x64 output tile per wave: 2 A-frags + 4 B-frags feed 8 WMMAs per K-step.
// ---------------------------------------------------------------------------
__global__ void k_qkv_gemm(const _Float16* __restrict__ xh,
                           const _Float16* __restrict__ WaT,
                           const float* __restrict__ b_attn,
                           _Float16* __restrict__ qh,
                           _Float16* __restrict__ kh,
                           _Float16* __restrict__ vh) {
  const int lane = threadIdx.x & 31;
  const int wave = threadIdx.x >> 5;
  const int m0 = blockIdx.x * 32;
  const int n0 = blockIdx.y * 256 + wave * 64;   // 64 cols = one head-block

  v8f acc[2][4];
#pragma unroll
  for (int rt = 0; rt < 2; ++rt)
#pragma unroll
    for (int ct = 0; ct < 4; ++ct)
      acc[rt][ct] = (v8f){0.f,0.f,0.f,0.f,0.f,0.f,0.f,0.f};

  for (int k0 = 0; k0 < C_DIM; k0 += 32) {
    v16h a0 = load_frag(xh, C_DIM, m0,      k0, lane);
    v16h a1 = load_frag(xh, C_DIM, m0 + 16, k0, lane);
#pragma unroll
    for (int ct = 0; ct < 4; ++ct) {
      v16h b = load_frag(WaT, C_DIM, n0 + ct * 16, k0, lane);
      acc[0][ct] = wmma_f16(a0, b, acc[0][ct]);
      acc[1][ct] = wmma_f16(a1, b, acc[1][ct]);
    }
  }

  // n0 is 64-aligned, so the whole 64-col tile lies in one head of one of q/k/v
  const int which = n0 / C_DIM;            // 0=q 1=k 2=v
  const int head  = (n0 % C_DIM) >> 6;
#pragma unroll
  for (int ct = 0; ct < 4; ++ct) {
    const int n    = n0 + ct * 16 + (lane & 15);
    const int dim  = ct * 16 + (lane & 15);  // n0%64 == 0
    const float bias = b_attn[n];
#pragma unroll
    for (int rt = 0; rt < 2; ++rt)
#pragma unroll
      for (int r = 0; r < 8; ++r) {
        const int m  = m0 + rt * 16 + r + ((lane >> 4) << 3);  // D-layout row
        const int b_ = m >> 11;                                // / T_SEQ
        const int t  = m & (T_SEQ - 1);
        const _Float16 val = (_Float16)(acc[rt][ct][r] + bias);
        const size_t bh = (size_t)(b_ * NHEAD + head);
        if (which == 0)
          qh[((bh * T_SEQ + t) << 6) + dim] = val;
        else if (which == 1)
          kh[((bh * T_SEQ + t) << 6) + dim] = val;
        else
          vh[((bh << 6) + dim) * T_SEQ + t] = val;             // transposed
      }
  }
}

// ---------------------------------------------------------------------------
// Chunk mean pooling: means[B*32][768]
// ---------------------------------------------------------------------------
__global__ void k_chunk_means(const float* __restrict__ x,
                              float* __restrict__ means) {
  const int bn = blockIdx.x;                 // b*32 + chunk
  const int b  = bn >> 5;
  const int nc = bn & 31;
  const float* xp = x + ((size_t)b * T_SEQ + nc * 64) * C_DIM;
  for (int c = threadIdx.x; c < C_DIM; c += blockDim.x) {
    float s = 0.f;
    for (int r = 0; r < 64; ++r) s += xp[(size_t)r * C_DIM + c];
    means[(size_t)bn * C_DIM + c] = s * (1.0f / 64.0f);
  }
}

// ---------------------------------------------------------------------------
// Routing embeds: re[B*32][32] = l2norm(means @ W_router + b_router)
// ---------------------------------------------------------------------------
__global__ void k_routing(const float* __restrict__ means,
                          const float* __restrict__ Wr,
                          const float* __restrict__ br,
                          float* __restrict__ re) {
  const int bn = blockIdx.x;
  const int r  = threadIdx.x;  // 0..31
  const float* mp = means + (size_t)bn * C_DIM;
  float acc = br[r];
  for (int k = 0; k < C_DIM; ++k) acc += mp[k] * Wr[k * 32 + r];
  float ss = acc * acc;
#pragma unroll
  for (int m = 16; m; m >>= 1) ss += __shfl_xor(ss, m, 32);
  re[bn * 32 + r] = acc / fmaxf(sqrtf(ss), 1e-12f);
}

// ---------------------------------------------------------------------------
// Gate + cosine scores + top-8 chunk membership bitmask per query token.
// One wave per (b,t). Matches jax.lax.top_k tie-breaking (lowest index first).
// ---------------------------------------------------------------------------
__global__ void k_gate_topk(const float* __restrict__ x,
                            const float* __restrict__ Wg,
                            const float* __restrict__ bg,
                            const float* __restrict__ re,
                            unsigned* __restrict__ selmask) {
  __shared__ float qr[32];
  __shared__ float sc[32];
  const int bt = blockIdx.x;           // b*T + t
  const int b  = bt >> 11;             // / T_SEQ
  const int r  = threadIdx.x;
  const float* xp = x + (size_t)bt * C_DIM;

  float acc = bg[r];
  for (int k = 0; k < C_DIM; ++k) acc += xp[k] * Wg[k * 32 + r];
  float ss = acc * acc;
#pragma unroll
  for (int m = 16; m; m >>= 1) ss += __shfl_xor(ss, m, 32);
  qr[r] = acc / fmaxf(sqrtf(ss), 1e-12f);
  __syncthreads();

  const float* reb = re + b * NCHUNK * 32;
  float s = 0.f;
  for (int k = 0; k < 32; ++k) s += qr[k] * reb[r * 32 + k];  // lane r = chunk r
  sc[r] = s;
  __syncthreads();

  if (r == 0) {
    unsigned mask = 0;
    for (int pass = 0; pass < 8; ++pass) {
      float best = -INFINITY;
      int bi = 0;
      for (int n = 0; n < NCHUNK; ++n)
        if (sc[n] > best) { best = sc[n]; bi = n; }
      mask |= 1u << bi;
      sc[bi] = -INFINITY;
    }
    selmask[bt] = mask;
  }
}

// ---------------------------------------------------------------------------
// Retrieval-gated sparse flash attention.
// Block = one (b,h, query chunk of 64 rows); 4 fully decoupled waves, each
// owning 16 rows with per-wave chunk skipping (no block syncs in the loop:
// V is pre-transposed in global, only the per-wave P tile touches LDS).
// ---------------------------------------------------------------------------
__global__ void k_attn(const _Float16* __restrict__ qh,
                       const _Float16* __restrict__ kh,
                       const _Float16* __restrict__ vh,   // [B,H,64,T]
                       const unsigned* __restrict__ selmask,
                       _Float16* __restrict__ yh) {
  __shared__ _Float16 lds_p[4][16 * 72];   // per-wave P tile, padded stride

  const int lane = threadIdx.x & 31;
  const int wave = threadIdx.x >> 5;
  const int hlf  = lane >> 4;              // 0/1: which M-half this lane holds
  const int jcol = lane & 15;

  const int qc = blockIdx.x;               // query chunk 0..31
  const int bh = blockIdx.y;               // 0..B*H-1
  const int b  = bh / NHEAD;
  const int h  = bh % NHEAD;

  const _Float16* qb = qh + (size_t)bh * T_SEQ * HD;
  const _Float16* kb = kh + (size_t)bh * T_SEQ * HD;
  const _Float16* vb = vh + (size_t)bh * HD * T_SEQ;   // transposed
  const int qrow0 = qc * 64 + wave * 16;

  // Q fragments for this wave's 16 rows (hd = 64 -> two K slices)
  v16h aq0 = load_frag(qb, HD, qrow0, 0, lane);
  v16h aq1 = load_frag(qb, HD, qrow0, 32, lane);

  // Selection bitmasks for the 8 rows this lane holds in D-layout
  unsigned um[8];
#pragma unroll
  for (int r = 0; r < 8; ++r)
    um[r] = selmask[b * T_SEQ + qrow0 + hlf * 8 + r];

  // Wave-level union of selected chunks (16 rows -> finest usable skip grain)
  unsigned u = 0;
#pragma unroll
  for (int r = 0; r < 8; ++r) u |= um[r];
#pragma unroll
  for (int m = 16; m; m >>= 1) u |= __shfl_xor(u, m, 32);

  float mrow[8], lrow[8];
  v8f o[4];
#pragma unroll
  for (int r = 0; r < 8; ++r) { mrow[r] = -INFINITY; lrow[r] = 0.f; }
#pragma unroll
  for (int t = 0; t < 4; ++t) o[t] = (v8f){0.f,0.f,0.f,0.f,0.f,0.f,0.f,0.f};

  _Float16* P = lds_p[wave];

  for (int jc = 0; jc <= qc; ++jc) {
    // Chunk-level sparsity: chunk touches this wave's 16 rows iff it overlaps
    // the local window of some row (jc*64+63 >= qrow0-255) or was retrieved.
    if (!((jc * 64 + 318 >= qrow0) || ((u >> jc) & 1u))) continue;

    // Prefetch next chunk's K rows and V^T segments (global_prefetch_b8);
    // hides L2 latency behind the 16 WMMAs below.
    if (jc < qc) {
      __builtin_prefetch(kb + ((size_t)((jc + 1) * 64 + lane * 2) << 6), 0, 1);
      __builtin_prefetch(vb + (size_t)(lane * 2) * T_SEQ + (jc + 1) * 64, 0, 1);
    }

    // ---- S = Q K^T over 64 keys (4 column tiles x 2 K slices) ----
    v8f s[4];
#pragma unroll
    for (int t = 0; t < 4; ++t) {
      v16h bk0 = load_frag(kb, HD, jc * 64 + t * 16, 0, lane);
      v16h bk1 = load_frag(kb, HD, jc * 64 + t * 16, 32, lane);
      v8f c = (v8f){0.f,0.f,0.f,0.f,0.f,0.f,0.f,0.f};
      c = wmma_f16(aq0, bk0, c);
      c = wmma_f16(aq1, bk1, c);
      s[t] = c;
    }

    // ---- scale + mask (causal & (local | retrieved)) ----
    const float scale = 0.125f;  // 1/sqrt(64)
#pragma unroll
    for (int t = 0; t < 4; ++t) {
      const int j = jc * 64 + t * 16 + jcol;
#pragma unroll
      for (int r = 0; r < 8; ++r) {
        const int i = qrow0 + hlf * 8 + r;
        const bool ok =
            (j <= i) && (((i - j) < LWIN) || ((um[r] >> jc) & 1u));
        s[t][r] = ok ? s[t][r] * scale : -1e30f;
      }
    }

    // ---- online softmax: row max / exp / row sum (16-lane half reductions) ----
    float nm[8], alpha[8], rsum[8];
#pragma unroll
    for (int r = 0; r < 8; ++r) {
      float rm = fmaxf(fmaxf(s[0][r], s[1][r]), fmaxf(s[2][r], s[3][r]));
#pragma unroll
      for (int m = 1; m <= 8; m <<= 1) rm = fmaxf(rm, __shfl_xor(rm, m, 32));
      nm[r]    = fmaxf(mrow[r], rm);
      alpha[r] = __expf(mrow[r] - nm[r]);
      mrow[r]  = nm[r];
      rsum[r]  = 0.f;
    }
#pragma unroll
    for (int t = 0; t < 4; ++t)
#pragma unroll
      for (int r = 0; r < 8; ++r) {
        const float p = __expf(s[t][r] - nm[r]);
        s[t][r] = p;
        rsum[r] += p;
      }
#pragma unroll
    for (int r = 0; r < 8; ++r) {
#pragma unroll
      for (int m = 1; m <= 8; m <<= 1) rsum[r] += __shfl_xor(rsum[r], m, 32);
      lrow[r] = lrow[r] * alpha[r] + rsum[r];
    }

    // ---- P (D-layout) -> per-wave LDS -> A-fragment layout ----
#pragma unroll
    for (int t = 0; t < 4; ++t)
#pragma unroll
      for (int r = 0; r < 8; ++r)
        P[(hlf * 8 + r) * 72 + t * 16 + jcol] = (_Float16)s[t][r];

    v16h ap0 = load_frag(P, 72, 0, 0, lane);
    v16h ap1 = load_frag(P, 72, 0, 32, lane);

    // ---- O = O*alpha + P V  (V^T read straight from global, b128 loads) ----
    const _Float16* vchunk = vb + jc * 64;
#pragma unroll
    for (int t = 0; t < 4; ++t) {
#pragma unroll
      for (int r = 0; r < 8; ++r) o[t][r] *= alpha[r];
      v16h bv0 = load_frag(vchunk, T_SEQ, t * 16, 0, lane);
      v16h bv1 = load_frag(vchunk, T_SEQ, t * 16, 32, lane);
      o[t] = wmma_f16(ap0, bv0, o[t]);
      o[t] = wmma_f16(ap1, bv1, o[t]);
    }
  }

  // ---- epilogue: normalize and emit y in [B,T,C] f16 for the proj GEMM ----
  float inv[8];
#pragma unroll
  for (int r = 0; r < 8; ++r) inv[r] = 1.0f / lrow[r];
#pragma unroll
  for (int t = 0; t < 4; ++t)
#pragma unroll
    for (int r = 0; r < 8; ++r) {
      const int i = qrow0 + hlf * 8 + r;
      yh[((size_t)b * T_SEQ + i) * C_DIM + h * HD + t * 16 + jcol] =
          (_Float16)(o[t][r] * inv[r]);
    }
}

// ---------------------------------------------------------------------------
// Output projection: yh[4096,768] @ WpT'[768,768] + b_proj -> out f32
// Same 32x64 tile-per-wave scheme as the QKV GEMM.
// ---------------------------------------------------------------------------
__global__ void k_proj_gemm(const _Float16* __restrict__ yh,
                            const _Float16* __restrict__ WpT,
                            const float* __restrict__ b_proj,
                            float* __restrict__ out) {
  const int lane = threadIdx.x & 31;
  const int wave = threadIdx.x >> 5;
  const int m0 = blockIdx.x * 32;
  const int n0 = blockIdx.y * 256 + wave * 64;

  v8f acc[2][4];
#pragma unroll
  for (int rt = 0; rt < 2; ++rt)
#pragma unroll
    for (int ct = 0; ct < 4; ++ct)
      acc[rt][ct] = (v8f){0.f,0.f,0.f,0.f,0.f,0.f,0.f,0.f};

  for (int k0 = 0; k0 < C_DIM; k0 += 32) {
    v16h a0 = load_frag(yh, C_DIM, m0,      k0, lane);
    v16h a1 = load_frag(yh, C_DIM, m0 + 16, k0, lane);
#pragma unroll
    for (int ct = 0; ct < 4; ++ct) {
      v16h b = load_frag(WpT, C_DIM, n0 + ct * 16, k0, lane);
      acc[0][ct] = wmma_f16(a0, b, acc[0][ct]);
      acc[1][ct] = wmma_f16(a1, b, acc[1][ct]);
    }
  }

#pragma unroll
  for (int ct = 0; ct < 4; ++ct) {
    const int n = n0 + ct * 16 + (lane & 15);
    const float bias = b_proj[n];
#pragma unroll
    for (int rt = 0; rt < 2; ++rt)
#pragma unroll
      for (int r = 0; r < 8; ++r) {
        const int m = m0 + rt * 16 + r + ((lane >> 4) << 3);
        out[(size_t)m * C_DIM + n] = acc[rt][ct][r] + bias;
      }
  }
}

// ---------------------------------------------------------------------------
// Host-side launcher
// ---------------------------------------------------------------------------
extern "C" void kernel_launch(void* const* d_in, const int* in_sizes, int n_in,
                              void* d_out, int out_size, void* d_ws,
                              size_t ws_size, hipStream_t stream) {
  const float* x        = (const float*)d_in[0];
  const float* W_attn   = (const float*)d_in[1];
  const float* b_attn   = (const float*)d_in[2];
  const float* W_proj   = (const float*)d_in[3];
  const float* b_proj   = (const float*)d_in[4];
  const float* W_router = (const float*)d_in[5];
  const float* b_router = (const float*)d_in[6];
  const float* W_gate   = (const float*)d_in[7];
  const float* b_gate   = (const float*)d_in[8];
  float* out = (float*)d_out;

  // workspace bump allocator (256B aligned)
  char* w = (char*)d_ws;
  auto alloc = [&](size_t bytes) -> void* {
    void* p = (void*)w;
    w += (bytes + 255) & ~(size_t)255;
    return p;
  };
  _Float16* xh   = (_Float16*)alloc((size_t)M_ROWS * C_DIM * 2);
  _Float16* WaT  = (_Float16*)alloc((size_t)N3C * C_DIM * 2);
  _Float16* WpT  = (_Float16*)alloc((size_t)C_DIM * C_DIM * 2);
  _Float16* qh   = (_Float16*)alloc((size_t)B_SZ * NHEAD * T_SEQ * HD * 2);
  _Float16* kh   = (_Float16*)alloc((size_t)B_SZ * NHEAD * T_SEQ * HD * 2);
  _Float16* vh   = (_Float16*)alloc((size_t)B_SZ * NHEAD * T_SEQ * HD * 2);
  _Float16* yh   = (_Float16*)alloc((size_t)M_ROWS * C_DIM * 2);
  float* means   = (float*)alloc((size_t)B_SZ * NCHUNK * C_DIM * 4);
  float* re      = (float*)alloc((size_t)B_SZ * NCHUNK * 32 * 4);
  unsigned* sel  = (unsigned*)alloc((size_t)M_ROWS * 4);

  // 1. precision casts / weight transposes
  {
    int n = M_ROWS * C_DIM;
    k_cast_f16<<<(n + 255) / 256, 256, 0, stream>>>(x, xh, n);
  }
  {
    int n = C_DIM * N3C;
    k_cast_transpose<<<(n + 255) / 256, 256, 0, stream>>>(W_attn, WaT, C_DIM, N3C);
  }
  {
    int n = C_DIM * C_DIM;
    k_cast_transpose<<<(n + 255) / 256, 256, 0, stream>>>(W_proj, WpT, C_DIM, C_DIM);
  }

  // 2. QKV projection (WMMA, 32x64 tile per wave; v written transposed)
  k_qkv_gemm<<<dim3(M_ROWS / 32, N3C / 256), 128, 0, stream>>>(
      xh, WaT, b_attn, qh, kh, vh);

  // 3. router / gate path (scalar; negligible FLOPs)
  k_chunk_means<<<B_SZ * NCHUNK, 256, 0, stream>>>(x, means);
  k_routing<<<B_SZ * NCHUNK, 32, 0, stream>>>(means, W_router, b_router, re);
  k_gate_topk<<<M_ROWS, 32, 0, stream>>>(x, W_gate, b_gate, re, sel);

  // 4. retrieval-gated sparse flash attention (WMMA, per-wave chunk skipping)
  k_attn<<<dim3(T_SEQ / 64, B_SZ * NHEAD), 128, 0, stream>>>(
      qh, kh, vh, sel, yh);

  // 5. output projection (WMMA)
  k_proj_gemm<<<dim3(M_ROWS / 32, C_DIM / 256), 128, 0, stream>>>(
      yh, WpT, b_proj, out);
}
